// LocationSlayer_6408091205834
// MI455X (gfx1250) — matrix operand order repaired and checked by reading, not compiled, not run on hardware
//
#include <hip/hip_runtime.h>
#include <hip/hip_bf16.h>

typedef __attribute__((ext_vector_type(16))) _Float16 v16h;
typedef __attribute__((ext_vector_type(8)))  _Float16 v8h;
typedef __attribute__((ext_vector_type(8)))  float    v8f;

#define THETA_F 10.0f

// CDNA5 async global->LDS copy path (ASYNCcnt), guarded so the file compiles
// on toolchains that lack the builtins (falls back to direct global gathers).
#if defined(__has_builtin)
#  if __has_builtin(__builtin_amdgcn_global_load_async_to_lds_b128) && \
      __has_builtin(__builtin_amdgcn_s_wait_asynccnt)
#    define USE_ASYNC_LDS 1
#  endif
#endif
#ifndef USE_ASYNC_LDS
#  define USE_ASYNC_LDS 0
#endif

// b128 async copy operand types: int4 pointers in AS1 (global) / AS3 (LDS).
typedef int v4i __attribute__((vector_size(16)));
typedef __attribute__((address_space(1))) v4i as1_v4i;
typedef __attribute__((address_space(3))) v4i as3_v4i;

// ---------------------------------------------------------------------------
// f32 -> f16 2-D pad-convert: dst (dstR x dstC), src (srcR x srcC), pad = 0
// ---------------------------------------------------------------------------
__global__ __launch_bounds__(256)
void pad2d_f32_f16(const float* __restrict__ src, _Float16* __restrict__ dst,
                   int srcR, int srcC, int dstC, int totalDst) {
  int i = blockIdx.x * blockDim.x + threadIdx.x;
  if (i >= totalDst) return;
  int r = i / dstC, c = i % dstC;
  dst[i] = (r < srcR && c < srcC) ? (_Float16)src[r * srcC + c] : (_Float16)0.0f;
}

// f32 -> f16 3-D pad-convert for x: (N,156,1000) -> (N,160,1024), pad = 0
__global__ __launch_bounds__(256)
void pad_x_f32_f16(const float* __restrict__ src, _Float16* __restrict__ dst,
                   int N, int C, int T, int CP, int TP) {
  size_t i = (size_t)blockIdx.x * blockDim.x + threadIdx.x;
  size_t total = (size_t)N * CP * TP;
  if (i >= total) return;
  int t = (int)(i % TP);
  int c = (int)((i / TP) % CP);
  int n = (int)(i / ((size_t)CP * TP));
  dst[i] = (c < C && t < T) ? (_Float16)src[((size_t)n * C + c) * T + t]
                            : (_Float16)0.0f;
}

// ---------------------------------------------------------------------------
// Batched WMMA GEMM:  D[n, m, j] = sum_k A[m,k] * B[n][k,j]
//   A : f16, M x Kpad row-major (weights, shared across batch), zero-padded
//   B : f16, element (n,k,j) at  B + n*bBatch + k*BK + j*bJ, zero-padded so
//       no bounds checks are needed (Kpad % 32 == 0, Jpad % 16 == 0)
//   D : f32, n*Dbatch + m*Jpad + j
// One wave computes MT vertically-stacked 16x16 tiles at one j-tile; the MT
// A fragments are preloaded as a batch, then MT WMMAs issue back-to-back
// against one B fragment (independent accumulators -> no hazards).
// BK == 1          : K-contiguous B -> direct b128 loads (branch-B GEMM).
// BK != 1 (async)  : 32x16 B slab DMA'd to LDS with 2x GLOBAL_LOAD_ASYNC_
//                    TO_LDS_B128 per slab, double-buffered against the WMMAs
//                    (s_wait_asynccnt 2 steady-state, 0 on the peeled tail).
// ---------------------------------------------------------------------------
template <int MT, int BK>
__global__ __launch_bounds__(256)
void wmma_gemm_f16(const _Float16* __restrict__ A, const _Float16* __restrict__ B,
                   float* __restrict__ D, int Kpad, int Jpad,
                   long bBatch, int bJ, long Dbatch, int tilesJ, int tilesTotal) {
  int wave = threadIdx.x >> 5;
  int lane = threadIdx.x & 31;
  int tile = blockIdx.x * 8 + wave;
  if (tile >= tilesTotal) return;           // wave-uniform exit, EXEC stays full
  int n   = blockIdx.y;
  int tsm = tile / tilesJ;                  // super-tile row (MT*16 rows)
  int tj  = tile % tilesJ;
  int m0  = tsm * (16 * MT);
  int j0  = tj * 16;

  int row = lane & 15;
  int hi  = lane >> 4;                      // 0: lanes 0-15, 1: lanes 16-31

  const _Float16* Arows[MT];
#pragma unroll
  for (int mt = 0; mt < MT; ++mt)
    Arows[mt] = A + (size_t)(m0 + mt * 16 + row) * Kpad;

  const _Float16* Bn = B + (size_t)n * bBatch;
  int j = j0 + row;

  const v8f vzero = {0.f, 0.f, 0.f, 0.f, 0.f, 0.f, 0.f, 0.f};
  v8f acc[MT];
#pragma unroll
  for (int mt = 0; mt < MT; ++mt) acc[mt] = vzero;

  // Preload all MT A fragments (batched loads, single wait), then issue the
  // MT WMMAs back-to-back.  A layout: element e -> K = k0+e+hi*16.
  auto do_mt_wmmas = [&](int k0, const v16h& b) {
    v16h a[MT];
#pragma unroll
    for (int mt = 0; mt < MT; ++mt) {
      const _Float16* ap = Arows[mt] + k0 + hi * 16;
      v8h alo = *(const v8h*)ap;
      v8h ahi = *(const v8h*)(ap + 8);
      a[mt] = __builtin_shufflevector(alo, ahi, 0, 1, 2, 3, 4, 5, 6, 7,
                                      8, 9, 10, 11, 12, 13, 14, 15);
    }
#pragma unroll
    for (int mt = 0; mt < MT; ++mt)
      acc[mt] = __builtin_amdgcn_wmma_f32_16x16x32_f16(
          false, a[mt], false, b, (short)0, acc[mt], false, false);
  };

  if constexpr (BK != 1 && USE_ASYNC_LDS) {
    // Per-wave double-buffered 32x16 B slab in LDS (32 rows x 32 bytes).
    __shared__ _Float16 btile[8][2][32 * 16];
    auto issue = [&](int k0, int buf) {
      // lane = K-row of the slab; each row = 32 contiguous bytes (bJ == 1)
      const _Float16* gp = Bn + (size_t)(k0 + lane) * BK + j0;
      _Float16* lp = &btile[wave][buf][lane * 16];
      __builtin_amdgcn_global_load_async_to_lds_b128(
          (as1_v4i*)(void*)gp, (as3_v4i*)(void*)lp, 0, 0);
      __builtin_amdgcn_global_load_async_to_lds_b128(
          (as1_v4i*)(void*)(gp + 8), (as3_v4i*)(void*)(lp + 8), 0, 0);
    };
    auto compute = [&](int k0, int buf) {
      const _Float16* lt = &btile[wave][buf][0];
      v16h b;
#pragma unroll
      for (int e = 0; e < 16; ++e)           // element e -> K = e + hi*16
        b[e] = lt[(e + hi * 16) * 16 + row];
      do_mt_wmmas(k0, b);
    };
    issue(0, 0);
    int buf = 0;
    int k0 = 0;
    for (; k0 + 32 < Kpad; k0 += 32, buf ^= 1) {
      issue(k0 + 32, buf ^ 1);               // overlap next slab's DMA
      __builtin_amdgcn_s_wait_asynccnt(2);   // oldest slab (2 ops) complete
      compute(k0, buf);
    }
    __builtin_amdgcn_s_wait_asynccnt(0);     // peeled tail: last slab landed
    compute(k0, buf);
  } else {
    for (int k0 = 0; k0 < Kpad; k0 += 32) {
      v16h b;
      if constexpr (BK == 1) {
        // B K-contiguous: element e -> K = k0 + e + hi*16; two b128 loads
        const _Float16* bp = Bn + (size_t)j * bJ + k0 + hi * 16;
        __builtin_prefetch(bp + 64, 0, 1);
        v8h blo = *(const v8h*)bp;
        v8h bhi = *(const v8h*)(bp + 8);
        b = __builtin_shufflevector(blo, bhi, 0, 1, 2, 3, 4, 5, 6, 7,
                                    8, 9, 10, 11, 12, 13, 14, 15);
      } else {
        // fallback: 16 immediate-offset gathers (wave-coalesced per element)
        const _Float16* bp = Bn + (size_t)(k0 + hi * 16) * BK + j;
        __builtin_prefetch(bp + (size_t)32 * BK, 0, 1);
#pragma unroll
        for (int e = 0; e < 16; ++e) b[e] = bp[(size_t)e * BK];
      }
      do_mt_wmmas(k0, b);
    }
  }

  float* Dn = D + (size_t)n * Dbatch;
#pragma unroll
  for (int mt = 0; mt < MT; ++mt) {
    int mb = m0 + mt * 16 + hi * 8;
#pragma unroll
    for (int r = 0; r < 8; ++r)
      Dn[(size_t)(mb + r) * Jpad + j] = acc[mt][r];
  }
}

// ---------------------------------------------------------------------------
// Fused PSP (77-tap causal alpha-kernel conv) + spike scan, one block per
// (n, channel) row.  U row stride = rowStride (padded); L = real length.
// sh_out (f16, stride shStride) gets spikes with ZERO-filled pad columns so
// the next GEMM can load unguarded.  f32_out writes into the concatenated
// (32,20,1156) output tensor at column offset tOff.
// ---------------------------------------------------------------------------
__global__ __launch_bounds__(256)
void conv_scan(const float* __restrict__ U, int Mrows, int L, int rowStride,
               _Float16* __restrict__ sh_out, int shStride,
               float* __restrict__ f32_out, int outRows, int outStride, int tOff) {
  __shared__ float sU[1000];
  __shared__ float sS[1000];
  __shared__ float sK[80];

  int row = blockIdx.x;
  int n = row / Mrows;
  int o = row % Mrows;
  int tid = threadIdx.x;

  // SRM alpha kernel: k/tau * exp(1 - k/tau), tau = 10, K = 77 taps
  for (int k = tid; k < 77; k += blockDim.x)
    sK[k] = (k * 0.1f) * __expf(1.0f - k * 0.1f);

  const float* Ur = U + (size_t)row * rowStride;
  for (int t = tid; t < L; t += blockDim.x) sU[t] = Ur[t];
  __syncthreads();

  // causal convolution, parallel over t
  for (int t = tid; t < L; t += blockDim.x) {
    float acc = 0.0f;
    int kmax = (t < 76) ? t : 76;
    for (int k = 0; k <= kmax; ++k) acc += sK[k] * sU[t - k];
    sS[t] = acc;
  }
  __syncthreads();

  // sequential spike scan (inherently serial per row): 10-tap refractory state
  if (tid == 0) {
    float ref[10];
#pragma unroll
    for (int i = 0; i < 10; ++i) ref[i] = 0.0f;
    float rk[10];
#pragma unroll
    for (int i = 0; i < 10; ++i) {
      float tt = (float)(i + 1);                 // REF_KERNEL[1..10]
      rk[i] = -20.0f * tt * __expf(1.0f - tt);   // -SCALE_REF*THETA * t * e^(1-t)
    }
    for (int t = 0; t < L; ++t) {
      float u = sS[t] + ref[0];
      float s = (u >= THETA_F) ? 1.0f : 0.0f;
#pragma unroll
      for (int i = 0; i < 9; ++i) ref[i] = ref[i + 1] + s * rk[i];
      ref[9] = s * rk[9];
      sS[t] = s;                                 // overwrite with spike
    }
  }
  __syncthreads();

  if (sh_out) {
    _Float16* dst = sh_out + (size_t)row * shStride;
    for (int t = tid; t < shStride; t += blockDim.x)
      dst[t] = (t < L) ? (_Float16)sS[t] : (_Float16)0.0f;   // zero pad columns
  }
  if (f32_out && o < outRows) {
    float* dst = f32_out + ((size_t)n * outRows + o) * outStride + tOff;
    for (int t = tid; t < L; t += blockDim.x) dst[t] = sS[t];
  }
}

// ---------------------------------------------------------------------------
// Orchestration
// ---------------------------------------------------------------------------
extern "C" void kernel_launch(void* const* d_in, const int* in_sizes, int n_in,
                              void* d_out, int out_size, void* d_ws, size_t ws_size,
                              hipStream_t stream) {
  (void)in_sizes; (void)n_in; (void)out_size; (void)ws_size;
  const float* x   = (const float*)d_in[0];   // (32,156,1,1,1000)
  const float* W1  = (const float*)d_in[1];   // (512,156)
  const float* W2  = (const float*)d_in[2];   // (20,512)
  const float* Wl1 = (const float*)d_in[3];   // (512,1000)
  const float* Wl2 = (const float*)d_in[4];   // (20,512)
  float* out = (float*)d_out;                 // (32,20,1,1,1156)

  const int N = 32, CIN = 156, HID = 512, COUT = 20, T = 1000;
  const int CINP = 160, TP = 1024, MP = 32;   // padded dims (K%32, J%16, M%32)

  char* ws = (char*)d_ws;
  size_t cur = 0;
  auto alloc = [&](size_t bytes) -> char* {
    char* p = ws + cur;
    cur += (bytes + 255) & ~(size_t)255;
    return p;
  };
  _Float16* xh   = (_Float16*)alloc((size_t)N * CINP * TP * 2);   // (32,160,1024)
  _Float16* W1h  = (_Float16*)alloc((size_t)HID * CINP * 2);      // (512,160)
  _Float16* W2h  = (_Float16*)alloc((size_t)MP * HID * 2);        // (32,512)
  _Float16* Wl1h = (_Float16*)alloc((size_t)HID * TP * 2);        // (512,1024)
  _Float16* Wl2h = (_Float16*)alloc((size_t)MP * HID * 2);        // (32,512)
  float*    U1   = (float*)   alloc((size_t)N * HID * TP * 4);    // (32,512,1024)
  _Float16* s1h  = (_Float16*)alloc((size_t)N * HID * TP * 2);    // (32,512,1024)
  float*    U2   = (float*)   alloc((size_t)N * MP * TP * 4);     // (32,32,1024)
  float*    U3   = (float*)   alloc((size_t)N * HID * CINP * 4);  // (32,512,160)
  _Float16* l1h  = (_Float16*)alloc((size_t)N * HID * CINP * 2);  // (32,512,160)
  float*    U4   = (float*)   alloc((size_t)N * MP * CINP * 4);   // (32,32,160)

  // --- zero-padded f16 conversions (spikes are exact in f16) ---
  {
    size_t total = (size_t)N * CINP * TP;
    pad_x_f32_f16<<<(unsigned)((total + 255) / 256), 256, 0, stream>>>(
        x, xh, N, CIN, T, CINP, TP);
  }
  pad2d_f32_f16<<<(HID * CINP + 255) / 256, 256, 0, stream>>>(
      W1, W1h, HID, CIN, CINP, HID * CINP);
  pad2d_f32_f16<<<(HID * TP + 255) / 256, 256, 0, stream>>>(
      Wl1, Wl1h, HID, T, TP, HID * TP);
  pad2d_f32_f16<<<(MP * HID + 255) / 256, 256, 0, stream>>>(
      W2, W2h, COUT, HID, HID, MP * HID);
  pad2d_f32_f16<<<(MP * HID + 255) / 256, 256, 0, stream>>>(
      Wl2, Wl2h, COUT, HID, HID, MP * HID);

  // --- Branch A ---
  {  // U1[n,o,t] = sum_c W1[o,c] * x[n,c,t] : M=512 K=160 J=1024, BK=TP
    int tilesJ = TP / 16;                       // 64
    int tilesTot = (HID / 64) * tilesJ;         // 8 * 64 = 512
    dim3 grid((tilesTot + 7) / 8, N);
    wmma_gemm_f16<4, TP><<<grid, 256, 0, stream>>>(
        W1h, xh, U1, CINP, TP, (long)CINP * TP, 1, (long)HID * TP,
        tilesJ, tilesTot);
  }
  conv_scan<<<N * HID, 256, 0, stream>>>(U1, HID, T, TP, s1h, TP,
                                         nullptr, 0, 0, 0);
  {  // U2[n,o,t] = sum_h W2[o,h] * s1[n,h,t] : M=32 K=512 J=1024, BK=TP
    int tilesJ = TP / 16;                       // 64
    int tilesTot = (MP / 32) * tilesJ;          // 64
    dim3 grid((tilesTot + 7) / 8, N);
    wmma_gemm_f16<2, TP><<<grid, 256, 0, stream>>>(
        W2h, s1h, U2, HID, TP, (long)HID * TP, 1, (long)MP * TP,
        tilesJ, tilesTot);
  }
  conv_scan<<<N * MP, 256, 0, stream>>>(U2, MP, T, TP, nullptr, 0,
                                        out, COUT, 1156, 0);

  // --- Branch B ---
  {  // U3[n,o,c] = sum_t Wl1[o,t] * x[n,c,t] : M=512 K=1024 J=160, BK=1
    int tilesJ = CINP / 16;                     // 10
    int tilesTot = (HID / 64) * tilesJ;         // 80
    dim3 grid((tilesTot + 7) / 8, N);
    wmma_gemm_f16<4, 1><<<grid, 256, 0, stream>>>(
        Wl1h, xh, U3, TP, CINP, (long)CINP * TP, TP, (long)HID * CINP,
        tilesJ, tilesTot);
  }
  conv_scan<<<N * HID, 256, 0, stream>>>(U3, HID, CIN, CINP, l1h, CINP,
                                         nullptr, 0, 0, 0);
  {  // U4[n,o,c] = sum_h Wl2[o,h] * l1[n,h,c] : M=32 K=512 J=160, BK=CINP
    int tilesJ = CINP / 16;                     // 10
    int tilesTot = (MP / 32) * tilesJ;          // 10
    dim3 grid((tilesTot + 7) / 8, N);
    wmma_gemm_f16<2, CINP><<<grid, 256, 0, stream>>>(
        Wl2h, l1h, U4, HID, CINP, (long)HID * CINP, 1, (long)MP * CINP,
        tilesJ, tilesTot);
  }
  conv_scan<<<N * MP, 256, 0, stream>>>(U4, MP, CIN, CINP, nullptr, 0,
                                        out, COUT, 1156, 1000);
}
